// OverlapGATv2_1597727834337
// MI455X (gfx1250) — matrix operand order repaired and checked by reading, not compiled, not run on hardware
//
#include <hip/hip_runtime.h>
#include <math.h>

// ---------------------------------------------------------------------------
// Types for CDNA5 WMMA (wave32, v_wmma_f32_16x16x32_f16)
// ---------------------------------------------------------------------------
typedef __attribute__((ext_vector_type(16))) _Float16 v16h;
typedef __attribute__((ext_vector_type(8)))  _Float16 v8h;
typedef __attribute__((ext_vector_type(8)))  float    v8f;

union AFrag { v16h v; v8h h8[2]; _Float16 e[16]; };

static __device__ __forceinline__ int imin(int a, int b) { return a < b ? a : b; }

// ---------------------------------------------------------------------------
// Generic TN GEMM:  C[m,n] = act( sum_k A[m,k]*B[n,k] + bias[n] )
// A element (m,k) at A[m*lda + k + bz*batA]          (fp16, K-contiguous)
// B element (k,n) at B[n*ldb + k + bz*batB]          (fp16, K-contiguous)
// C element (m,n) at C[m*rsC + n*csC + bz*batC]      (fp32)
// One wave = one 16x16 tile via v_wmma_f32_16x16x32_f16.
// Fragment layout per CDNA5 ISA 7.12.2 (16-bit A 16x32 / B 32x16):
//   A lane (m%16)+16h: elems e[0..7] = K 8h+0..7, e[8..15] = K 16+8h+0..7
//     -> two contiguous 8-half runs  (two b128 loads)
//   B lane (n%16)+16h: elems e[0..7] = K 8h+0..7, e[8..15] = K 16+8h+0..7
//     -> two contiguous 8-half runs  (two b128 loads)
// Out-of-range rows/cols are CLAMPED for loads (each C element is an
// independent dot product, so clamped lanes only affect never-stored outputs);
// the K tail uses unconditional clamped loads + select-to-zero (branch-free).
// ---------------------------------------------------------------------------
__global__ __launch_bounds__(32) void gemm_f16_wmma(
    const _Float16* __restrict__ A, const _Float16* __restrict__ Bm,
    const float* __restrict__ bias, float* __restrict__ C,
    int M, int N, int K,
    long lda, long batA, long ldb, long batB,
    long rsC, long csC, long batC, int act)
{
  const int lane = threadIdx.x;
  const int tn = blockIdx.x << 4;
  const int tm = blockIdx.y << 4;
  const int bz = blockIdx.z;
  const _Float16* Ab = A + batA * bz;
  const _Float16* Bb = Bm + batB * bz;
  float* Cb = C + batC * bz;

  const int l15 = lane & 15;
  const int h   = lane >> 4;
  const int amc = imin(tm + l15, M - 1);   // clamped row for loads
  const int bnc = imin(tn + l15, N - 1);   // clamped col for loads
  const long abase = (long)amc * lda;
  const long bbase = (long)bnc * ldb;

  // fast path requires 16-byte aligned runs: lda, ldb multiples of 8 halves
  const int kFast = (((lda | ldb) & 7) == 0) ? (K & ~31) : 0;

  v8f acc = {};
  for (int kt = 0; kt < kFast; kt += 32) {
    AFrag a, b;
    const _Float16* ap = Ab + abase + kt + (h << 3);
    a.h8[0] = *(const v8h*)(ap);
    a.h8[1] = *(const v8h*)(ap + 16);
    const _Float16* bp = Bb + bbase + kt + (h << 3);
    b.h8[0] = *(const v8h*)(bp);
    b.h8[1] = *(const v8h*)(bp + 16);
    if (kt + 64 < kFast) {                  // uniform-guarded prefetch
      __builtin_prefetch(ap + 64, 0, 1);
      __builtin_prefetch(bp + 64, 0, 1);
    }
    acc = __builtin_amdgcn_wmma_f32_16x16x32_f16(false, a.v, false, b.v,
                                                 (short)0, acc, false, false);
  }
  for (int kt = kFast; kt < K; kt += 32) {  // branch-free guarded tail
    AFrag a, b;
#pragma unroll
    for (int v = 0; v < 8; ++v) {
      int ka0 = kt + (h << 3) + (v & 3) * 2 + ((v >> 2) << 4);
      int ka1 = ka0 + 1;
      _Float16 va0 = Ab[abase + imin(ka0, K - 1)];
      _Float16 va1 = Ab[abase + imin(ka1, K - 1)];
      a.e[2*v]   = (ka0 < K) ? va0 : (_Float16)0.f;
      a.e[2*v+1] = (ka1 < K) ? va1 : (_Float16)0.f;
      int kb0 = kt + (h << 3) + (v & 3) * 2 + ((v >> 2) << 4);
      int kb1 = kb0 + 1;
      _Float16 vb0 = Bb[bbase + imin(kb0, K - 1)];
      _Float16 vb1 = Bb[bbase + imin(kb1, K - 1)];
      b.e[2*v]   = (kb0 < K) ? vb0 : (_Float16)0.f;
      b.e[2*v+1] = (kb1 < K) ? vb1 : (_Float16)0.f;
    }
    acc = __builtin_amdgcn_wmma_f32_16x16x32_f16(false, a.v, false, b.v,
                                                 (short)0, acc, false, false);
  }
#pragma unroll
  for (int v = 0; v < 8; ++v) {
    int m = tm + v + (h << 3);
    int n = tn + l15;
    if (m < M && n < N) {
      float val = acc[v];
      if (bias) val += bias[n];
      if (act == 1) val = val > 0.f ? val : 0.f;
      Cb[(long)m * rsC + (long)n * csC] = val;
    }
  }
}

// ---------------------------------------------------------------------------
// Direct conv (kernel (KH,1), stride (S,1), VALID) + ReLU, fp32.
// ---------------------------------------------------------------------------
__global__ void conv_relu(const float* __restrict__ in, const float* __restrict__ wgt,
                          float* __restrict__ out, int Bn, int Cin, int Hin,
                          int Cout, int Hout, int W, int KH, int S)
{
  long idx = blockIdx.x * (long)blockDim.x + threadIdx.x;
  long total = (long)Bn * Cout * Hout * W;
  if (idx >= total) return;
  int w = (int)(idx % W); long t = idx / W;
  int ho = (int)(t % Hout); t /= Hout;
  int co = (int)(t % Cout); int b = (int)(t / Cout);
  float acc = 0.f;
  for (int ci = 0; ci < Cin; ++ci) {
    const float* ip = in + ((long)(b * Cin + ci) * Hin + ho * S) * W + w;
    const float* wp = wgt + ((long)co * Cin + ci) * KH;
    for (int k = 0; k < KH; ++k) acc += ip[(long)k * W] * wp[k];
  }
  out[idx] = acc > 0.f ? acc : 0.f;
}

__global__ void cvt_f32_to_f16(const float* __restrict__ in, _Float16* __restrict__ out, long n)
{
  long i = blockIdx.x * (long)blockDim.x + threadIdx.x;
  if (i < n) out[i] = (_Float16)in[i];
}

// w[Kd,Nd] f32 -> wt[Nd,Kd] f16  (K-contiguous for GEMM B operand)
__global__ void tcvt_f32_to_f16(const float* __restrict__ w, _Float16* __restrict__ wt,
                                int Kd, int Nd)
{
  long i = blockIdx.x * (long)blockDim.x + threadIdx.x;
  long total = (long)Kd * Nd;
  if (i >= total) return;
  int n = (int)(i % Nd); int k = (int)(i / Nd);
  wt[(long)n * Kd + k] = (_Float16)w[i];
}

// hidden_w[(f*64+k)*256 + n] f32 -> wt[n*65536 + k*1024 + f] f16
// (transpose + fold the VLAD [F,K]->flat permutation; K-contiguous)
__global__ void hperm_cvt(const float* __restrict__ w, _Float16* __restrict__ wt)
{
  long j = blockIdx.x * (long)blockDim.x + threadIdx.x;
  if (j >= 16777216L) return;
  int n = (int)(j >> 16);
  int r = (int)(j & 65535);
  int k = r >> 10; int f = r & 1023;
  wt[j] = (_Float16)w[((long)(f * 64 + k) << 8) + n];
}

// [B,C,W] fp32 -> [B,W,C] fp16  (conv6 output -> node-major)
__global__ void t_cw_to_wc_h(const float* __restrict__ in, _Float16* __restrict__ out,
                             int Bn, int Cc, int W)
{
  long i = blockIdx.x * (long)blockDim.x + threadIdx.x;
  long total = (long)Bn * Cc * W;
  if (i >= total) return;
  int w = (int)(i % W); long t = i / W;
  int c = (int)(t % Cc); int b = (int)(t / Cc);
  out[((long)b * W + w) * Cc + c] = (_Float16)in[i];
}

// ---------------------------------------------------------------------------
// GAT edge kernels
// ---------------------------------------------------------------------------
__global__ void csr_count(const int* __restrict__ dst, int* __restrict__ cnt, int E, int Nn)
{
  int n = blockIdx.x * blockDim.x + threadIdx.x;
  if (n >= Nn) return;
  int c = 0;
  for (int e = 0; e < E; ++e) c += (dst[e] == n);
  cnt[n] = c;
}

__global__ void csr_scan(const int* __restrict__ cnt, int* __restrict__ rows, int Nn)
{
  if (blockIdx.x == 0 && threadIdx.x == 0) {
    int a = 0; rows[0] = 0;
    for (int n = 0; n < Nn; ++n) { a += cnt[n]; rows[n + 1] = a; }
  }
}

__global__ void csr_fill(const int* __restrict__ dst, const int* __restrict__ rows,
                         int* __restrict__ cols, int E, int Nn)
{
  int n = blockIdx.x * blockDim.x + threadIdx.x;
  if (n >= Nn) return;
  int p = rows[n];
  for (int e = 0; e < E; ++e) if (dst[e] == n) cols[p++] = e;
}

__global__ void gat_edge_score(const float* __restrict__ xl, const float* __restrict__ xr,
                               const float* __restrict__ att, const int* __restrict__ src,
                               const int* __restrict__ dst, float* __restrict__ score,
                               unsigned* __restrict__ smax, int E, int Bn, int Nn, int D)
{
  int t = blockIdx.x * blockDim.x + threadIdx.x;
  if (t >= E * Bn) return;
  int b = t % Bn; int e = t / Bn;
  int s = src[e], d = dst[e];
  const float* pl = xl + ((long)b * Nn + s) * D;
  const float* pr = xr + ((long)b * Nn + d) * D;
  float acc = 0.f;
  for (int i = 0; i < D; ++i) {
    float m = pl[i] + pr[i];
    m = m > 0.f ? m : 0.2f * m;          // leaky_relu 0.2
    acc += m * att[i];
  }
  score[t] = acc;
  unsigned u = __float_as_uint(acc);
  u = (u & 0x80000000u) ? ~u : (u | 0x80000000u);  // monotone key
  atomicMax(&smax[(long)d * Bn + b], u);
}

__global__ void gat_edge_exp(float* __restrict__ score, const int* __restrict__ dst,
                             const unsigned* __restrict__ smax, float* __restrict__ den,
                             int E, int Bn)
{
  int t = blockIdx.x * blockDim.x + threadIdx.x;
  if (t >= E * Bn) return;
  int b = t % Bn; int e = t / Bn;
  int d = dst[e];
  unsigned k = smax[(long)d * Bn + b];
  unsigned u = (k & 0x80000000u) ? (k & 0x7FFFFFFFu) : ~k;
  float mx = __uint_as_float(u);
  float ex = __expf(score[t] - mx);
  score[t] = ex;
  atomicAdd(&den[(long)d * Bn + b], ex);
}

// out (= x@Wres + bias) += sum_e alpha_e * xl[b, src_e, :]   per (b, dst)
__global__ void gat_aggregate(const float* __restrict__ xl, const float* __restrict__ exsc,
                              const float* __restrict__ den, const int* __restrict__ rows,
                              const int* __restrict__ cols, const int* __restrict__ src,
                              float* __restrict__ out, int Bn, int Nn, int D)
{
  const int n = blockIdx.x % Nn;
  const int b = blockIdx.x / Nn;
  const int tid = threadIdx.x;               // 128
  const int e0 = rows[n], e1 = rows[n + 1];
  const float inv = 1.f / den[(long)n * Bn + b];
  __shared__ float salpha[64];
  __shared__ int   ssrc[64];
  float acc[4] = {0.f, 0.f, 0.f, 0.f};       // D <= 512
  const int nd = (D + blockDim.x - 1) / blockDim.x;
  for (int ce = e0; ce < e1; ce += 64) {
    int cn = e1 - ce; if (cn > 64) cn = 64;
    if (tid < cn) {
      int e = cols[ce + tid];
      salpha[tid] = exsc[(long)e * Bn + b] * inv;
      ssrc[tid] = src[e];
    }
    __syncthreads();
    for (int i = 0; i < cn; ++i) {
      float al = salpha[i];
      const float* p = xl + ((long)b * Nn + ssrc[i]) * D;
      for (int j = 0; j < nd; ++j) {
        int d = tid + j * blockDim.x;
        if (d < D) acc[j] += al * p[d];
      }
    }
    __syncthreads();
  }
  float* po = out + ((long)b * Nn + n) * D;
  for (int j = 0; j < nd; ++j) {
    int d = tid + j * blockDim.x;
    if (d < D) po[d] += acc[j];
  }
}

// ---------------------------------------------------------------------------
// L2 normalize each node row of [B*S, F] (F=1024 contiguous); emit:
//   feat_h  [B*S, F]            f16 (K-contiguous A for assignment GEMM)
//   feat_t  [B, F, FT_LD]       f16 (K-contiguous B for vlad GEMM, padded ld)
// ---------------------------------------------------------------------------
#define FT_LD 912
__global__ void l2norm_rows_dual(const float* __restrict__ in, _Float16* __restrict__ feat_h,
                                 _Float16* __restrict__ feat_t, int Sn, int Fn)
{
  int r = blockIdx.x;                        // r = b*Sn + s
  int s = r % Sn; int b = r / Sn;
  int t = threadIdx.x;                       // 256
  const float* p = in + (long)r * Fn;
  __shared__ float red[256];
  float ss = 0.f;
  for (int f = t; f < Fn; f += blockDim.x) { float v = p[f]; ss += v * v; }
  red[t] = ss; __syncthreads();
  for (int st = 128; st > 0; st >>= 1) { if (t < st) red[t] += red[t + st]; __syncthreads(); }
  float inv = 1.f / fmaxf(sqrtf(red[0]), 1e-12f);
  for (int f = t; f < Fn; f += blockDim.x) {
    _Float16 v = (_Float16)(p[f] * inv);
    feat_h[(long)r * Fn + f] = v;
    feat_t[((long)b * Fn + f) * FT_LD + s] = v;
  }
}

// softmax over 64 clusters per row; emit f32 in place and transposed f16
// act_t [B, 64, FT_LD]
__global__ void softmax64(float* __restrict__ x, _Float16* __restrict__ act_t, int Sn)
{
  int r = blockIdx.x; int t = threadIdx.x;   // blockDim == 64
  int s = r % Sn; int b = r / Sn;
  __shared__ float red[64];
  float v = x[(long)r * 64 + t];
  red[t] = v; __syncthreads();
  for (int st = 32; st > 0; st >>= 1) { if (t < st) red[t] = fmaxf(red[t], red[t + st]); __syncthreads(); }
  float mx = red[0]; __syncthreads();
  float ex = __expf(v - mx);
  red[t] = ex; __syncthreads();
  for (int st = 32; st > 0; st >>= 1) { if (t < st) red[t] += red[t + st]; __syncthreads(); }
  float o = ex / red[0];
  x[(long)r * 64 + t] = o;
  act_t[((long)b * 64 + t) * FT_LD + s] = (_Float16)o;
}

__global__ void asum_k(const float* __restrict__ act, float* __restrict__ a_sum,
                       int Bn, int Sn, int Kc)
{
  int t = blockIdx.x * blockDim.x + threadIdx.x;
  if (t >= Bn * Kc) return;
  int k = t % Kc; int b = t / Kc;
  float s = 0.f;
  for (int ss = 0; ss < Sn; ++ss) s += act[((long)b * Sn + ss) * Kc + k];
  a_sum[t] = s;
}

// vlad_t[b,k,f] -= a_sum[b,k]*cw2[f,k];  then intra-normalize over f per (b,k)
__global__ void vlad_fix_intranorm(float* __restrict__ vlad_t, const float* __restrict__ a_sum,
                                   const float* __restrict__ cw2, int Fn, int Kc)
{
  int k = blockIdx.x % Kc; int b = blockIdx.x / Kc;
  int t = threadIdx.x;                       // 256
  float as = a_sum[b * Kc + k];
  float* row = vlad_t + ((long)b * Kc + k) * Fn;
  __shared__ float red[256];
  float ss = 0.f;
  for (int f = t; f < Fn; f += blockDim.x) {
    float v = row[f] - as * cw2[(long)f * Kc + k];
    row[f] = v;
    ss += v * v;
  }
  red[t] = ss; __syncthreads();
  for (int st = 128; st > 0; st >>= 1) { if (t < st) red[t] += red[t + st]; __syncthreads(); }
  float inv = 1.f / fmaxf(sqrtf(red[0]), 1e-12f);
  for (int f = t; f < Fn; f += blockDim.x) row[f] *= inv;
}

__global__ void l2flat_h(const float* __restrict__ in, _Float16* __restrict__ out, long n)
{
  int b = blockIdx.x; int t = threadIdx.x;   // 256
  __shared__ float red[256];
  const float* p = in + (long)b * n;
  float ss = 0.f;
  for (long i = t; i < n; i += blockDim.x) { float v = p[i]; ss += v * v; }
  red[t] = ss; __syncthreads();
  for (int st = 128; st > 0; st >>= 1) { if (t < st) red[t] += red[t + st]; __syncthreads(); }
  float inv = 1.f / fmaxf(sqrtf(red[0]), 1e-12f);
  for (long i = t; i < n; i += blockDim.x) out[(long)b * n + i] = (_Float16)(p[i] * inv);
}

__global__ void bn2_apply(float* __restrict__ hid, _Float16* __restrict__ hid_h,
                          const float* __restrict__ g, const float* __restrict__ be,
                          int Bn, int On)
{
  int t = blockIdx.x * blockDim.x + threadIdx.x;
  if (t >= Bn * On) return;
  int n = t % On;
  float v = hid[t] * (1.f / sqrtf(1.f + 1e-5f)) * g[n] + be[n];
  hid[t] = v;
  hid_h[t] = (_Float16)v;
}

__global__ void gate_final(const float* __restrict__ hid, const float* __restrict__ gpre,
                           float* __restrict__ out, int On)
{
  int b = blockIdx.x; int t = threadIdx.x;   // blockDim == On == 256
  float g = 1.f / (1.f + __expf(-gpre[(long)b * On + t]));
  float v = hid[(long)b * On + t] * g;
  __shared__ float red[256];
  red[t] = v * v; __syncthreads();
  for (int st = 128; st > 0; st >>= 1) { if (t < st) red[t] += red[t + st]; __syncthreads(); }
  float inv = 1.f / fmaxf(sqrtf(red[0]), 1e-12f);
  out[(long)b * On + t] = v * inv;
}

// ---------------------------------------------------------------------------
// Host orchestration
// ---------------------------------------------------------------------------
extern "C" void kernel_launch(void* const* d_in, const int* in_sizes, int n_in,
                              void* d_out, int out_size, void* d_ws, size_t ws_size,
                              hipStream_t stream)
{
  (void)n_in; (void)out_size; (void)ws_size;
  const int Bn = 4, Wn = 900, Nn = 900;

  const float* x_l = (const float*)d_in[0];
  const float* wconv[13];                              // w0..w11, wLast2
  for (int i = 0; i < 13; ++i) wconv[i] = (const float*)d_in[1 + i];
  struct GP { const float *Wl, *bl, *Wr, *br, *att, *Wres, *bias; };
  GP gp[3];
  int dins[3] = {256, 256, 512}, douts[3] = {256, 512, 512};
  for (int l = 0; l < 3; ++l) {
    int base = 14 + 7 * l;
    gp[l].Wl = (const float*)d_in[base + 0]; gp[l].bl = (const float*)d_in[base + 1];
    gp[l].Wr = (const float*)d_in[base + 2]; gp[l].br = (const float*)d_in[base + 3];
    gp[l].att = (const float*)d_in[base + 4]; gp[l].Wres = (const float*)d_in[base + 5];
    gp[l].bias = (const float*)d_in[base + 6];
  }
  const float* cluster_w  = (const float*)d_in[35];
  const float* cluster_b  = (const float*)d_in[36];
  const float* cluster_w2 = (const float*)d_in[37];
  const float* hidden_w   = (const float*)d_in[38];
  const float* bn2_g      = (const float*)d_in[39];
  const float* bn2_b      = (const float*)d_in[40];
  const float* gate_w     = (const float*)d_in[41];
  const float* gate_b     = (const float*)d_in[42];
  const int*   esrc       = (const int*)d_in[43];
  const int*   edst       = (const int*)d_in[44];
  const int E = in_sizes[43];

  // bump allocator over d_ws
  char* wsb = (char*)d_ws; size_t off = 0;
  auto alloc = [&](size_t bytes) -> void* {
    off = (off + 255) & ~(size_t)255;
    void* p = wsb + off; off += bytes; return p;
  };
  float*    ping0  = (float*)alloc(3456000UL * 4);
  float*    ping1  = (float*)alloc(3456000UL * 4);
  float*    gemA   = (float*)alloc(921600UL * 4);
  float*    gemB   = (float*)alloc(921600UL * 4);
  _Float16* h0     = (_Float16*)alloc(921600UL * 2);
  _Float16* h1     = (_Float16*)alloc(921600UL * 2);
  _Float16* wbuf   = (_Float16*)alloc(524288UL * 2);
  _Float16* xh     = (_Float16*)alloc(1843200UL * 2);
  float*    xlb    = (float*)alloc(1843200UL * 4);
  float*    xrb    = (float*)alloc(1843200UL * 4);
  float*    resb   = (float*)alloc(1843200UL * 4);
  float*    score  = (float*)alloc((size_t)E * Bn * 4);
  float*    den    = (float*)alloc(3600UL * 4);
  unsigned* smax   = (unsigned*)alloc(3600UL * 4);
  int*      cnt    = (int*)alloc(904UL * 4);
  int*      rows   = (int*)alloc(904UL * 4);
  int*      cols   = (int*)alloc((size_t)E * 4);
  float*    featcw = (float*)alloc(3686400UL * 4);
  _Float16* feat_h = (_Float16*)alloc(3686400UL * 2);
  _Float16* feat_t = (_Float16*)alloc((size_t)Bn * 1024 * FT_LD * 2);
  float*    assign = (float*)alloc(230400UL * 4);
  _Float16* act_t  = (_Float16*)alloc((size_t)Bn * 64 * FT_LD * 2);
  float*    a_sum  = (float*)alloc(256UL * 4);
  float*    vlad_t = (float*)alloc(262144UL * 4);
  _Float16* vlad_h = (_Float16*)alloc(262144UL * 2);
  _Float16* hwh    = (_Float16*)alloc(16777216UL * 2);
  float*    hid    = (float*)alloc(1024UL * 4);
  _Float16* hid_h  = (_Float16*)alloc(1024UL * 2);
  float*    gpre   = (float*)alloc(1024UL * 4);

  auto CVT = [&](const float* src, _Float16* dst, long n) {
    cvt_f32_to_f16<<<dim3((unsigned)((n + 255) / 256)), 256, 0, stream>>>(src, dst, n);
  };
  auto TCVT = [&](const float* src, _Float16* dst, int Kd, int Nd) {
    long n = (long)Kd * Nd;
    tcvt_f32_to_f16<<<dim3((unsigned)((n + 255) / 256)), 256, 0, stream>>>(src, dst, Kd, Nd);
  };
  auto GEMM = [&](const _Float16* A, const _Float16* Bm, const float* bias, float* C,
                  int M, int N, int K, long lda, long batA, long ldb, long batB,
                  long rsC, long csC, long batC, int act, int nbat) {
    dim3 g((unsigned)((N + 15) / 16), (unsigned)((M + 15) / 16), (unsigned)nbat);
    gemm_f16_wmma<<<g, 32, 0, stream>>>(A, Bm, bias, C, M, N, K, lda, batA,
                                        ldb, batB, rsC, csC, batC, act);
  };

  // ---- CSR of in-edges per destination node (deterministic) ----
  csr_count<<<dim3((Nn + 127) / 128), 128, 0, stream>>>(edst, cnt, E, Nn);
  csr_scan<<<1, 1, 0, stream>>>(cnt, rows, Nn);
  csr_fill<<<dim3((Nn + 127) / 128), 128, 0, stream>>>(edst, rows, cols, E, Nn);

  // ---- conv0..conv6 (direct fp32) ----
  struct CS { int Cin, Cout, KH, S, Hin, Hout; };
  CS cs[7] = {{5,16,5,1,64,60},{16,32,3,2,60,29},{32,64,3,2,29,14},{64,64,3,2,14,6},
              {64,128,2,2,6,3},{128,128,1,2,3,2},{128,128,1,2,2,1}};
  const float* cin = x_l;
  float* pp[2] = {ping0, ping1};
  for (int i = 0; i < 7; ++i) {
    float* o = pp[i & 1];
    long tot = (long)Bn * cs[i].Cout * cs[i].Hout * Wn;
    conv_relu<<<dim3((unsigned)((tot + 255) / 256)), 256, 0, stream>>>(
        cin, wconv[i], o, Bn, cs[i].Cin, cs[i].Hin, cs[i].Cout, cs[i].Hout, Wn, cs[i].KH, cs[i].S);
    cin = o;
  }
  // conv6 out [4,128,1,900] channel-major -> node-major f16 [4,900,128]
  t_cw_to_wc_h<<<dim3((unsigned)((460800 + 255) / 256)), 256, 0, stream>>>(cin, h0, Bn, 128, Wn);

  // ---- conv7..conv10: TN GEMM, node-major in/out, relu ----
  _Float16* hin = h0; _Float16* hout = h1;
  for (int i = 7; i <= 10; ++i) {
    CVT(wconv[i], wbuf, 128L * 128);                        // [Cout,Cin] K-contiguous
    GEMM(wbuf, hin, nullptr, gemA, 128, 900, 128,
         128, 0, 128, 900L * 128, 1, 128, 900L * 128, 1, Bn);  // store node-major
    CVT(gemA, hout, 460800L);
    _Float16* t = hin; hin = hout; hout = t;
  }
  // ---- conv11 (convLast1): [256,128], node-major out -> GAT input directly ----
  CVT(wconv[11], wbuf, 256L * 128);
  GEMM(wbuf, hin, nullptr, gemB, 256, 900, 128,
       128, 0, 128, 900L * 128, 1, 256, 900L * 256, 1, Bn);
  CVT(gemB, xh, 921600L);                                   // xh = [3600, 256]

  // ---- GATv2 layers ----
  for (int l = 0; l < 3; ++l) {
    int din = dins[l], dout = douts[l];
    TCVT(gp[l].Wl, wbuf, din, dout);                        // -> [dout, din]
    GEMM(xh, wbuf, gp[l].bl, xlb, 3600, dout, din, din, 0, din, 0, dout, 1, 0, 0, 1);
    TCVT(gp[l].Wr, wbuf, din, dout);
    GEMM(xh, wbuf, gp[l].br, xrb, 3600, dout, din, din, 0, din, 0, dout, 1, 0, 0, 1);
    TCVT(gp[l].Wres, wbuf, din, dout);
    GEMM(xh, wbuf, gp[l].bias, resb, 3600, dout, din, din, 0, din, 0, dout, 1, 0, 0, 1);

    hipMemsetAsync(smax, 0, 3600 * 4, stream);
    hipMemsetAsync(den, 0, 3600 * 4, stream);
    int eb = E * Bn;
    gat_edge_score<<<dim3((unsigned)((eb + 255) / 256)), 256, 0, stream>>>(
        xlb, xrb, gp[l].att, esrc, edst, score, smax, E, Bn, Nn, dout);
    gat_edge_exp<<<dim3((unsigned)((eb + 255) / 256)), 256, 0, stream>>>(
        score, edst, smax, den, E, Bn);
    gat_aggregate<<<dim3((unsigned)(Bn * Nn)), 128, 0, stream>>>(
        xlb, score, den, rows, cols, esrc, resb, Bn, Nn, dout);
    CVT(resb, xh, 3600L * dout);                            // next input (and convLast2 B)
  }

  // ---- convLast2: [1024,512] x nodes[900,512]^T per batch, relu, node-major out ----
  CVT(wconv[12], wbuf, 1024L * 512);
  GEMM(wbuf, xh, nullptr, featcw, 1024, 900, 512,
       512, 0, 512, 900L * 512, 1, 1024, 900L * 1024, 1, Bn);

  // ---- per-node channel L2 norm -> feat_h [3600,1024] + feat_t [B,1024,FT_LD] ----
  l2norm_rows_dual<<<dim3((unsigned)(Bn * Wn)), 256, 0, stream>>>(featcw, feat_h, feat_t, Wn, 1024);

  // ---- NetVLAD assignment: [3600,1024] x [1024,64] + bias, softmax ----
  TCVT(cluster_w, wbuf, 1024, 64);                          // -> [64, 1024]
  GEMM(feat_h, wbuf, cluster_b, assign, 3600, 64, 1024, 1024, 0, 1024, 0, 64, 1, 0, 0, 1);
  softmax64<<<dim3((unsigned)3600), 64, 0, stream>>>(assign, act_t, Wn);
  asum_k<<<1, 256, 0, stream>>>(assign, a_sum, Bn, Wn, 64);

  // ---- vlad^T[b,k,f] = sum_s act[s,k]*feat[s,f] : M=64,N=1024,K=900 per batch ----
  GEMM(act_t, feat_t, nullptr, vlad_t, 64, 1024, 900,
       FT_LD, 64L * FT_LD, FT_LD, 1024L * FT_LD, 1024, 1, 65536L, 0, Bn);
  vlad_fix_intranorm<<<dim3((unsigned)(Bn * 64)), 256, 0, stream>>>(vlad_t, a_sum, cluster_w2, 1024, 64);
  l2flat_h<<<dim3((unsigned)Bn), 256, 0, stream>>>(vlad_t, vlad_h, 65536L);

  // ---- hidden: [4,65536] x permuted hidden_w^T [256,65536] ----
  hperm_cvt<<<dim3((unsigned)(16777216L / 256)), 256, 0, stream>>>(hidden_w, hwh);
  GEMM(vlad_h, hwh, nullptr, hid, 4, 256, 65536, 65536, 0, 65536, 0, 256, 1, 0, 0, 1);
  bn2_apply<<<dim3((unsigned)4), 256, 0, stream>>>(hid, hid_h, bn2_g, bn2_b, Bn, 256);

  // ---- gating: [4,256] x [256,256]^T + bias, sigmoid*x, L2 norm -> out ----
  TCVT(gate_w, wbuf, 256, 256);
  GEMM(hid_h, wbuf, gate_b, gpre, 4, 256, 256, 256, 0, 256, 0, 256, 1, 0, 0, 1);
  gate_final<<<dim3((unsigned)Bn), 256, 0, stream>>>(hid, gpre, (float*)d_out, 256);
}
